// SequentialVQVAE_59648505807422
// MI455X (gfx1250) — matrix operand order repaired and checked by reading, not compiled, not run on hardware
//
#include <hip/hip_runtime.h>
#include <hip/hip_bf16.h>

typedef __attribute__((ext_vector_type(16))) _Float16 v16h;
typedef __attribute__((ext_vector_type(8)))  float    v8f;

#define NB    64      // batch
#define SEQ   64      // usable seq (65-1)
#define NIMG  4096    // NB*SEQ
#define NPIX  200704  // NIMG*49
#define KCONV 1152    // 9*128
#define KCHNK 36      // KCONV/32
#define CCH   128
#define NS    8       // slots
#define NK    512     // codes
#define NE    64      // code dim
#define DECAYF 0.99f
#define EPSF   1e-5f

// ---------------- WMMA fragment helpers (layouts per CDNA5 ISA 7.12.2) ----

// A 16x32 f16 fragment from an LDS tile stored row-major (row stride `stride` halves).
// lane 0-15: M=lane,   K = {k0+0..7, k0+16..23} ; lane16-31: M=lane-16, K = {k0+8..15, k0+24..31}
__device__ inline v16h load_a_lds(const _Float16* At, int stride, int lane, int k0) {
  v16h a;
  int m  = lane & 15;
  int kb = (lane >> 4) * 8;
#pragma unroll
  for (int e = 0; e < 8; ++e) {
    a[e]     = At[m * stride + k0 + kb + e];
    a[8 + e] = At[m * stride + k0 + 16 + kb + e];
  }
  return a;
}

// ---------------- small utility kernels ----------------------------------------------

// Pack row-major f32 [K x N] weights into per-fragment f16 order:
// dst[((ntile*(K/32)+kc)*32 + lane)*16 + e] with lane = ((kk/8)%2)*16 + n%16, e = kk%8 + (kk/16)*8.
// A B-fragment is then ONE contiguous 32-byte load per lane.
__global__ void pack_b_kernel(const float* __restrict__ src, _Float16* __restrict__ dst,
                              int K, int N) {
  int i = blockIdx.x * 256 + threadIdx.x;
  if (i >= K * N) return;
  int k = i / N, n = i % N;
  int ntile = n >> 4, nl = n & 15;
  int kc = k >> 5, kk = k & 31;
  int lane = ((kk >> 3) & 1) * 16 + nl;
  int e = (kk & 7) + ((kk >> 4) << 3);
  dst[(((size_t)ntile * (K >> 5) + kc) * 32 + lane) * 16 + e] = (_Float16)src[i];
}

__global__ void init_state_kernel(const float* __restrict__ codebook,
                                  float* cb, float* ea, float* h, float* cs, float* acc) {
  int i = blockIdx.x * 256 + threadIdx.x;
  if (i < NS * NK * NE) { cb[i] = codebook[i]; ea[i] = codebook[i]; }
  if (i < NB * 64)      h[i]  = 0.f;
  if (i < NS * NK)      cs[i] = 0.f;
  if (i < 2)            acc[i] = 0.f;
}

// embedding sum: x[img,pos,c] = sum_i emb[obs[b,s,i,pos] + 147*i][c]  (f16 out)
__global__ void embed_kernel(const int* __restrict__ obs, const float* __restrict__ emb,
                             _Float16* __restrict__ x) {
  int gid = blockIdx.x * 256 + threadIdx.x;
  if (gid >= NIMG * 49 * CCH) return;
  int c   = gid & 127;
  int pos = (gid >> 7) % 49;
  int img = gid / (49 * CCH);
  int b = img >> 6, s = img & 63;
  const int* op = obs + (size_t)((b * 65 + s) * 3) * 49 + pos;
  float acc = 0.f;
#pragma unroll
  for (int i = 0; i < 3; ++i) acc += emb[(size_t)(op[i * 49] + 147 * i) * CCH + c];
  x[gid] = (_Float16)acc;
}

// ---------------- conv as implicit GEMM with WMMA ------------------------------------
// Full 16x1152 im2col A tile staged ONCE in LDS in A-fragment order (36 KB),
// then 36 barrier-free WMMAs per wave; B from packed weights (one v16h load each).
__global__ void conv_wmma_kernel(const _Float16* __restrict__ xin,
                                 const _Float16* __restrict__ wBp,
                                 const float* __restrict__ bias,
                                 float* __restrict__ out) {
  __shared__ __align__(32) _Float16 At[KCHNK * 32 * 16];   // 18432 halves = 36 KB
  int m0 = blockIdx.x * 16;
  int tid = threadIdx.x, w = tid >> 5, lane = tid & 31;

  // stage im2col gather directly into per-lane fragment order
  for (int i = tid; i < KCHNK * 32 * 16; i += 256) {
    int e  = i & 15;
    int l  = (i >> 4) & 31;
    int kc = i >> 9;
    int m  = l & 15, kg = l >> 4;
    int kk = (e < 8) ? (kg * 8 + e) : (16 + kg * 8 + (e - 8));
    int k  = kc * 32 + kk;
    int p = m0 + m;
    int img = p / 49, pos = p % 49;
    int oy = pos / 7, ox = pos % 7;
    int tap = k >> 7, cin = k & 127;
    int iy = oy + tap / 3 - 1, ix = ox + tap % 3 - 1;
    _Float16 v = (_Float16)0.f;
    if ((unsigned)iy < 7u && (unsigned)ix < 7u)
      v = xin[((size_t)img * 49 + iy * 7 + ix) * CCH + cin];
    At[i] = v;
  }
  __syncthreads();

  const v16h* Ap = (const v16h*)At;
  const v16h* Bp = (const v16h*)wBp;
  v8f c = {};
#pragma unroll
  for (int kc = 0; kc < KCHNK; ++kc) {
    v16h a = Ap[kc * 32 + lane];
    v16h b = Bp[((size_t)w * KCHNK + kc) * 32 + lane];
    c = __builtin_amdgcn_wmma_f32_16x16x32_f16(false, a, false, b, (short)0, c, false, false);
  }
  int n = w * 16 + (lane & 15);
  int mb = (lane >> 4) * 8;
#pragma unroll
  for (int r = 0; r < 8; ++r)
    out[(size_t)(m0 + mb + r) * CCH + n] = c[r] + bias[n];
}

// ---------------- batchnorm ----------------------------------------------------------

__global__ void bn_stats_kernel(const float* __restrict__ x, float* mean, float* var) {
  __shared__ float s1[256], s2[256];
  int c = blockIdx.x, tid = threadIdx.x;
  float a1 = 0.f, a2 = 0.f;
  for (int p = tid; p < NPIX; p += 256) {
    float v = x[(size_t)p * CCH + c];
    a1 += v; a2 += v * v;
  }
  s1[tid] = a1; s2[tid] = a2;
  __syncthreads();
  for (int off = 128; off > 0; off >>= 1) {
    if (tid < off) { s1[tid] += s1[tid + off]; s2[tid] += s2[tid + off]; }
    __syncthreads();
  }
  if (tid == 0) {
    float m = s1[0] / (float)NPIX;
    mean[c] = m;
    var[c]  = s2[0] / (float)NPIX - m * m;
  }
}

__global__ void bn_apply_kernel(const float* __restrict__ x, const float* __restrict__ mean,
                                const float* __restrict__ var, const float* __restrict__ g,
                                const float* __restrict__ b, _Float16* __restrict__ y) {
  size_t i = (size_t)blockIdx.x * 256 + threadIdx.x;
  if (i >= (size_t)NPIX * CCH) return;
  int c = (int)(i & 127);
  float v = (x[i] - mean[c]) * rsqrtf(var[c] + EPSF) * g[c] + b[c];
  y[i] = (_Float16)(v > 0.f ? v : 0.f);
}

// max over spatial then relu -> obs_feat (f32)
__global__ void pool_kernel(const _Float16* __restrict__ x, float* __restrict__ obs_feat) {
  int img = blockIdx.x, c = threadIdx.x;
  float mx = -3.4e38f;
  for (int p = 0; p < 49; ++p)
    mx = fmaxf(mx, (float)x[((size_t)img * 49 + p) * CCH + c]);
  obs_feat[(size_t)img * CCH + c] = mx > 0.f ? mx : 0.f;
}

// ---------------- per-step kernels ---------------------------------------------------

// GRU step + z = h @ pre_w + pre_b. Single block of 256.
__global__ void gru_step_kernel(const float* __restrict__ obs_feat,
                                const float* __restrict__ wih, const float* __restrict__ whh,
                                const float* __restrict__ bih, const float* __restrict__ bhh,
                                const float* __restrict__ pre_w, const float* __restrict__ pre_b,
                                float* __restrict__ h, float* __restrict__ z, int t) {
  __shared__ float hold[NB * 64];
  __shared__ float hnew[NB * 64];
  for (int i = threadIdx.x; i < NB * 64; i += 256) hold[i] = h[i];
  __syncthreads();
  for (int i = threadIdx.x; i < NB * 64; i += 256) {
    int b = i >> 6, j = i & 63;
    const float* xt = obs_feat + (size_t)(b * SEQ + t) * CCH;
    float ir = bih[j], iz = bih[64 + j], in = bih[128 + j];
    for (int c = 0; c < CCH; ++c) {
      float xv = xt[c];
      ir += xv * wih[(size_t)j * CCH + c];
      iz += xv * wih[(size_t)(64 + j) * CCH + c];
      in += xv * wih[(size_t)(128 + j) * CCH + c];
    }
    float hr = bhh[j], hz = bhh[64 + j], hn = bhh[128 + j];
    for (int c = 0; c < 64; ++c) {
      float hv = hold[b * 64 + c];
      hr += hv * whh[j * 64 + c];
      hz += hv * whh[(64 + j) * 64 + c];
      hn += hv * whh[(128 + j) * 64 + c];
    }
    float r  = 1.f / (1.f + __expf(-(ir + hr)));
    float zz = 1.f / (1.f + __expf(-(iz + hz)));
    float nn = tanhf(in + r * hn);
    hnew[i] = (1.f - zz) * nn + zz * hold[i];
  }
  __syncthreads();
  for (int i = threadIdx.x; i < NB * 64; i += 256) h[i] = hnew[i];
  for (int i = threadIdx.x; i < NB * NS * NE; i += 256) {
    int b = i >> 9, j = i & 511;
    float acc = pre_b[j];
    for (int c = 0; c < 64; ++c) acc += hnew[b * 64 + c] * pre_w[c * 512 + j];
    z[i] = acc;
  }
}

// VQ nearest code per (b, slot): argmin over 512 codes, write z_q, accumulate e-loss.
__global__ void vq_assign_kernel(const float* __restrict__ z, const float* __restrict__ cb,
                                 float* __restrict__ zq, int* __restrict__ ind,
                                 float* __restrict__ acc) {
  __shared__ float zv[NE];
  __shared__ float sd[128];
  __shared__ int   sk[128];
  __shared__ float se[NE];
  int b = blockIdx.x >> 3, s = blockIdx.x & 7;
  int tid = threadIdx.x;
  if (tid < NE) zv[tid] = z[(size_t)(b * NS + s) * NE + tid];
  __syncthreads();
  float best = 3.4e38f; int bk = 0x7fffffff;
  for (int k = tid; k < NK; k += 128) {
    const float* cr = cb + (size_t)(s * NK + k) * NE;
    float d = 0.f;
    for (int e = 0; e < NE; ++e) { float df = zv[e] - cr[e]; d += df * df; }
    if (d < best || (d == best && k < bk)) { best = d; bk = k; }
  }
  sd[tid] = best; sk[tid] = bk;
  __syncthreads();
  for (int off = 64; off > 0; off >>= 1) {
    if (tid < off) {
      float d2 = sd[tid + off]; int k2 = sk[tid + off];
      if (d2 < sd[tid] || (d2 == sd[tid] && k2 < sk[tid])) { sd[tid] = d2; sk[tid] = k2; }
    }
    __syncthreads();
  }
  int kbest = sk[0];
  if (tid == 0) ind[b * NS + s] = kbest;
  if (tid < NE) {
    float cv = cb[(size_t)(s * NK + kbest) * NE + tid];
    zq[(size_t)(b * NS + s) * NE + tid] = cv;
    float df = cv - zv[tid];
    se[tid] = df * df;
  }
  __syncthreads();
  if (tid == 0) {
    float sum = 0.f;
    for (int e = 0; e < NE; ++e) sum += se[e];
    atomicAdd(&acc[1], sum * (1.0f / (float)(NB * NS * NE)));
  }
}

// EMA update of cs / ea per (slot, code). 64 threads = one per dim (and per batch for match).
__global__ void ema_update_kernel(const int* __restrict__ ind, const float* __restrict__ z,
                                  float* __restrict__ cs, float* __restrict__ ea) {
  __shared__ int match[NB];
  int s = blockIdx.x >> 9, k = blockIdx.x & 511;
  int e = threadIdx.x;
  match[e] = (ind[e * NS + s] == k) ? 1 : 0;   // thread e doubles as batch index
  __syncthreads();
  float sum = 0.f; int cnt = 0;
  for (int b = 0; b < NB; ++b)
    if (match[b]) { cnt++; sum += z[(size_t)(b * NS + s) * NE + e]; }
  size_t base = (size_t)(s * NK + k) * NE;
  ea[base + e] = ea[base + e] * DECAYF + (1.f - DECAYF) * sum;
  if (e == 0) cs[s * NK + k] = cs[s * NK + k] * DECAYF + (1.f - DECAYF) * (float)cnt;
}

__global__ void ema_ntot_kernel(const float* __restrict__ cs, float* __restrict__ ntot) {
  __shared__ float sm[256];
  int s = blockIdx.x, tid = threadIdx.x;
  sm[tid] = cs[s * NK + tid] + cs[s * NK + 256 + tid];
  __syncthreads();
  for (int off = 128; off > 0; off >>= 1) {
    if (tid < off) sm[tid] += sm[tid + off];
    __syncthreads();
  }
  if (tid == 0) ntot[s] = sm[0];
}

__global__ void ema_cb_kernel(const float* __restrict__ cs, const float* __restrict__ ea,
                              const float* __restrict__ ntot, float* __restrict__ cb) {
  int s = blockIdx.x >> 9, k = blockIdx.x & 511;
  float nt = ntot[s];
  float csn = (cs[s * NK + k] + EPSF) / (nt + (float)NK * EPSF) * nt;
  int e = threadIdx.x;
  size_t base = (size_t)(s * NK + k) * NE;
  cb[base + e] = ea[base + e] / csn;
}

// FiLM MLP: zq(64x512)@mlp1 -> relu -> @mlp2 -> gamma/beta. Single block.
__global__ void mlp_film_kernel(const float* __restrict__ zq,
                                const float* __restrict__ w1, const float* __restrict__ b1,
                                const float* __restrict__ w2, const float* __restrict__ b2,
                                float* __restrict__ gamma, float* __restrict__ beta) {
  __shared__ float hid[NB * CCH];
  for (int i = threadIdx.x; i < NB * CCH; i += 256) {
    int b = i >> 7, c = i & 127;
    const float* zr = zq + (size_t)b * 512;
    float acc = b1[c];
    for (int k = 0; k < 512; ++k) acc += zr[k] * w1[(size_t)k * CCH + c];
    hid[i] = acc > 0.f ? acc : 0.f;
  }
  __syncthreads();
  for (int i = threadIdx.x; i < NB * 256; i += 256) {
    int b = i >> 8, j = i & 255;
    float acc = b2[j];
    for (int k = 0; k < CCH; ++k) acc += hid[b * CCH + k] * w2[(size_t)k * 256 + j];
    if (j < 128) gamma[b * CCH + j] = 1.f + acc;
    else         beta[b * CCH + (j - 128)] = acc;
  }
}

// Fused FiLM -> dec1 -> dec2 -> dec3 -> log_softmax -> masked CE, WMMA GEMMs (packed B).
__global__ void decoder_step_kernel(const float* __restrict__ obs_feat,
                                    const float* __restrict__ gamma, const float* __restrict__ beta,
                                    const _Float16* __restrict__ w1p, const float* __restrict__ b1,
                                    const _Float16* __restrict__ w2p, const float* __restrict__ b2,
                                    const float* __restrict__ w3, const float* __restrict__ b3,
                                    const int* __restrict__ actions,
                                    float* __restrict__ acc, int t) {
  __shared__ _Float16 FA[16 * CCH];
  __shared__ _Float16 HB[16 * CCH];
  __shared__ _Float16 HC[16 * CCH];
  __shared__ float    LG[16 * 6];
  int r0 = blockIdx.x * 16;
  int tid = threadIdx.x, w = tid >> 5, lane = tid & 31;
  int n = w * 16 + (lane & 15);
  int mb = (lane >> 4) * 8;
  const v16h* B1 = (const v16h*)w1p;
  const v16h* B2 = (const v16h*)w2p;

  // stage relu(FiLM(obs_feat)) as f16
  for (int i = tid; i < 16 * CCH; i += 256) {
    int rr = i >> 7, c = i & 127;
    int row = r0 + rr, b = row >> 6;
    float v = obs_feat[(size_t)row * CCH + c] * gamma[b * CCH + c] + beta[b * CCH + c];
    FA[i] = (_Float16)(v > 0.f ? v : 0.f);
  }
  __syncthreads();

  // GEMM1: (16x128) @ dec1(128x128)
  v8f c1 = {};
#pragma unroll
  for (int kc = 0; kc < 4; ++kc) {
    v16h a = load_a_lds(FA, CCH, lane, kc * 32);
    v16h b = B1[(w * 4 + kc) * 32 + lane];
    c1 = __builtin_amdgcn_wmma_f32_16x16x32_f16(false, a, false, b, (short)0, c1, false, false);
  }
#pragma unroll
  for (int r = 0; r < 8; ++r) {
    float v = c1[r] + b1[n];
    HB[(mb + r) * CCH + n] = (_Float16)(v > 0.f ? v : 0.f);
  }
  __syncthreads();

  // GEMM2: @ dec2(128x128)
  v8f c2 = {};
#pragma unroll
  for (int kc = 0; kc < 4; ++kc) {
    v16h a = load_a_lds(HB, CCH, lane, kc * 32);
    v16h b = B2[(w * 4 + kc) * 32 + lane];
    c2 = __builtin_amdgcn_wmma_f32_16x16x32_f16(false, a, false, b, (short)0, c2, false, false);
  }
#pragma unroll
  for (int r = 0; r < 8; ++r) {
    float v = c2[r] + b2[n];
    HC[(mb + r) * CCH + n] = (_Float16)(v > 0.f ? v : 0.f);
  }
  __syncthreads();

  // dec3 (128->6), log_softmax, masked CE
  if (tid < 96) {
    int rr = tid / 6, j = tid % 6;
    float s = b3[j];
    for (int c = 0; c < CCH; ++c) s += (float)HC[rr * CCH + c] * w3[c * 6 + j];
    LG[rr * 6 + j] = s;
  }
  __syncthreads();
  if (tid < 16) {
    int row = r0 + tid, b = row >> 6, s = row & 63;
    float mx = LG[tid * 6];
    for (int j = 1; j < 6; ++j) mx = fmaxf(mx, LG[tid * 6 + j]);
    float se = 0.f;
    for (int j = 0; j < 6; ++j) se += __expf(LG[tid * 6 + j] - mx);
    float lse = mx + __logf(se);
    int a = (s < SEQ - 1) ? actions[b * (SEQ - 1) + s] : 0;
    float ce = lse - LG[tid * 6 + a];
    if (s >= t && s <= SEQ - 2) atomicAdd(&acc[0], ce);
  }
}

__global__ void finalize_kernel(const float* __restrict__ acc, float* __restrict__ out) {
  if (threadIdx.x == 0) { out[0] = acc[0] / (float)NB; out[1] = acc[1] / (float)NB; }
}

// ---------------- host-side orchestration --------------------------------------------

extern "C" void kernel_launch(void* const* d_in, const int* in_sizes, int n_in,
                              void* d_out, int out_size, void* d_ws, size_t ws_size,
                              hipStream_t stream) {
  (void)in_sizes; (void)n_in; (void)out_size; (void)ws_size;
  const int*   obs      = (const int*)  d_in[0];
  const int*   actions  = (const int*)  d_in[1];
  const float* emb      = (const float*)d_in[2];
  const float* conv1_w  = (const float*)d_in[3];
  const float* conv1_b  = (const float*)d_in[4];
  const float* bn1_g    = (const float*)d_in[5];
  const float* bn1_b    = (const float*)d_in[6];
  const float* conv2_w  = (const float*)d_in[7];
  const float* conv2_b  = (const float*)d_in[8];
  const float* bn2_g    = (const float*)d_in[9];
  const float* bn2_b    = (const float*)d_in[10];
  const float* gru_wih  = (const float*)d_in[11];
  const float* gru_whh  = (const float*)d_in[12];
  const float* gru_bih  = (const float*)d_in[13];
  const float* gru_bhh  = (const float*)d_in[14];
  const float* pre_w    = (const float*)d_in[15];
  const float* pre_b    = (const float*)d_in[16];
  const float* codebook = (const float*)d_in[17];
  const float* mlp1_w   = (const float*)d_in[18];
  const float* mlp1_b   = (const float*)d_in[19];
  const float* mlp2_w   = (const float*)d_in[20];
  const float* mlp2_b   = (const float*)d_in[21];
  const float* dec1_w   = (const float*)d_in[22];
  const float* dec1_b   = (const float*)d_in[23];
  const float* dec2_w   = (const float*)d_in[24];
  const float* dec2_b   = (const float*)d_in[25];
  const float* dec3_w   = (const float*)d_in[26];
  const float* dec3_b   = (const float*)d_in[27];
  float* out = (float*)d_out;

  char* ws = (char*)d_ws;
  size_t off = 0;
  auto alloc = [&](size_t bytes) -> void* {
    void* p = ws + off;
    off = (off + bytes + 255) & ~(size_t)255;
    return p;
  };
  _Float16* xf16    = (_Float16*)alloc((size_t)NIMG * 49 * CCH * 2);  // conv1 in / bn2 out (reused)
  float*    convout = (float*)   alloc((size_t)NPIX * CCH * 4);       // conv1 out, then conv2 out
  _Float16* bn1f16  = (_Float16*)alloc((size_t)NIMG * 49 * CCH * 2);  // conv2 in
  float*    obsfeat = (float*)   alloc((size_t)NIMG * CCH * 4);
  _Float16* c1p     = (_Float16*)alloc((size_t)KCONV * CCH * 2);      // packed B weights
  _Float16* c2p     = (_Float16*)alloc((size_t)KCONV * CCH * 2);
  _Float16* d1p     = (_Float16*)alloc((size_t)CCH * CCH * 2);
  _Float16* d2p     = (_Float16*)alloc((size_t)CCH * CCH * 2);
  float*    mean1   = (float*)alloc(CCH * 4);
  float*    var1    = (float*)alloc(CCH * 4);
  float*    mean2   = (float*)alloc(CCH * 4);
  float*    var2    = (float*)alloc(CCH * 4);
  float*    hbuf    = (float*)alloc((size_t)NB * 64 * 4);
  float*    zbuf    = (float*)alloc((size_t)NB * NS * NE * 4);
  float*    zqbuf   = (float*)alloc((size_t)NB * NS * NE * 4);
  int*      indbuf  = (int*)  alloc((size_t)NB * NS * 4);
  float*    cbw     = (float*)alloc((size_t)NS * NK * NE * 4);
  float*    eaw     = (float*)alloc((size_t)NS * NK * NE * 4);
  float*    csw     = (float*)alloc((size_t)NS * NK * 4);
  float*    ntot    = (float*)alloc(NS * 4);
  float*    gammab  = (float*)alloc((size_t)NB * CCH * 4);
  float*    betab   = (float*)alloc((size_t)NB * CCH * 4);
  float*    accb    = (float*)alloc(2 * 4);

  // init state + weight packing into B-fragment order
  init_state_kernel<<<(NS * NK * NE + 255) / 256, 256, 0, stream>>>(codebook, cbw, eaw, hbuf, csw, accb);
  pack_b_kernel<<<(KCONV * CCH + 255) / 256, 256, 0, stream>>>(conv1_w, c1p, KCONV, CCH);
  pack_b_kernel<<<(KCONV * CCH + 255) / 256, 256, 0, stream>>>(conv2_w, c2p, KCONV, CCH);
  pack_b_kernel<<<(CCH * CCH + 255) / 256, 256, 0, stream>>>(dec1_w, d1p, CCH, CCH);
  pack_b_kernel<<<(CCH * CCH + 255) / 256, 256, 0, stream>>>(dec2_w, d2p, CCH, CCH);

  // encoder
  embed_kernel<<<(NIMG * 49 * CCH + 255) / 256, 256, 0, stream>>>(obs, emb, xf16);
  conv_wmma_kernel<<<NPIX / 16, 256, 0, stream>>>(xf16, c1p, conv1_b, convout);
  bn_stats_kernel<<<CCH, 256, 0, stream>>>(convout, mean1, var1);
  bn_apply_kernel<<<(int)(((size_t)NPIX * CCH + 255) / 256), 256, 0, stream>>>(convout, mean1, var1, bn1_g, bn1_b, bn1f16);
  conv_wmma_kernel<<<NPIX / 16, 256, 0, stream>>>(bn1f16, c2p, conv2_b, convout);
  bn_stats_kernel<<<CCH, 256, 0, stream>>>(convout, mean2, var2);
  bn_apply_kernel<<<(int)(((size_t)NPIX * CCH + 255) / 256), 256, 0, stream>>>(convout, mean2, var2, bn2_g, bn2_b, xf16);
  pool_kernel<<<NIMG, CCH, 0, stream>>>(xf16, obsfeat);

  // sequential scan
  for (int t = 0; t < SEQ - 1; ++t) {
    gru_step_kernel<<<1, 256, 0, stream>>>(obsfeat, gru_wih, gru_whh, gru_bih, gru_bhh,
                                           pre_w, pre_b, hbuf, zbuf, t);
    vq_assign_kernel<<<NB * NS, 128, 0, stream>>>(zbuf, cbw, zqbuf, indbuf, accb);
    ema_update_kernel<<<NS * NK, NB, 0, stream>>>(indbuf, zbuf, csw, eaw);
    ema_ntot_kernel<<<NS, 256, 0, stream>>>(csw, ntot);
    ema_cb_kernel<<<NS * NK, NE, 0, stream>>>(csw, eaw, ntot, cbw);
    mlp_film_kernel<<<1, 256, 0, stream>>>(zqbuf, mlp1_w, mlp1_b, mlp2_w, mlp2_b, gammab, betab);
    decoder_step_kernel<<<NIMG / 16, 256, 0, stream>>>(obsfeat, gammab, betab,
                                                       d1p, dec1_b, d2p, dec2_b,
                                                       dec3_w, dec3_b, actions, accb, t);
  }
  finalize_kernel<<<1, 32, 0, stream>>>(accb, out);
}
